// MambaG2G_57595511439750
// MI455X (gfx1250) — compile-verified, matrix operand order checked by reading, not run on hardware
//
#include <hip/hip_runtime.h>
#include <hip/hip_bf16.h>
#include <math.h>

// ---------------------------------------------------------------------------
// Feature probes (CDNA5 data movers)
// ---------------------------------------------------------------------------
#if defined(__has_builtin)
# if __has_builtin(__builtin_amdgcn_tensor_load_to_lds) && __has_builtin(__builtin_amdgcn_s_wait_tensorcnt)
#  define USE_TDM 1
# endif
# if __has_builtin(__builtin_amdgcn_global_load_async_to_lds_b128) && __has_builtin(__builtin_amdgcn_s_wait_asynccnt)
#  define USE_ASYNC 1
# endif
#endif
#ifndef USE_TDM
# define USE_TDM 0
#endif
#ifndef USE_ASYNC
# define USE_ASYNC 0
#endif

// ---------------------------------------------------------------------------
// Types
// ---------------------------------------------------------------------------
typedef __bf16 bf16_t;
typedef __attribute__((ext_vector_type(16))) __bf16       v16bf;
typedef __attribute__((ext_vector_type(8)))  __bf16       v8bf;
typedef __attribute__((ext_vector_type(8)))  float        v8f;
typedef __attribute__((ext_vector_type(4)))  unsigned int u32x4;
typedef __attribute__((ext_vector_type(8)))  int          i32x8;
typedef __attribute__((ext_vector_type(4)))  int          i32x4;
typedef int v4i_t __attribute__((vector_size(16)));   // payload type of async-LDS builtins

// async global->LDS b128: (global src, lds dst, imm offset, imm cpol)
#define GASYNC_B128(gp, lp) \
    __builtin_amdgcn_global_load_async_to_lds_b128((v4i_t*)(gp), (v4i_t*)(lp), 0, 0)

#define DEV __device__ __forceinline__

DEV unsigned short f2bf_u(float f) {
    unsigned int u = __builtin_bit_cast(unsigned int, f);
    u += 0x7fffu + ((u >> 16) & 1u);           // round-to-nearest-even
    return (unsigned short)(u >> 16);
}
DEV bf16_t f2bf(float f) { return __builtin_bit_cast(bf16_t, f2bf_u(f)); }
DEV float bf2f(bf16_t h) {
    unsigned int u = ((unsigned int)__builtin_bit_cast(unsigned short, h)) << 16;
    return __builtin_bit_cast(float, u);
}

#if USE_TDM
// ---------------------------------------------------------------------------
// TDM: issue one 2D tile load (bf16 elements) global -> LDS.
//   tensor: tensor_d0 (row length, elems) x tensor_d1 (rows), row stride = stride0
//   tile:   tile_d0 x tile_d1, written to lds_addr with +16B pad per 64B row
//   (pad_interval=16 DWORDs, pad_amount=4 DWORDs -> 80B LDS row stride)
// Descriptor layout per CDNA5 ISA.
// ---------------------------------------------------------------------------
DEV void tdm_load_2d_bf16(unsigned lds_addr, const void* gaddr,
                          unsigned tensor_d0, unsigned tensor_d1,
                          unsigned tile_d0, unsigned tile_d1,
                          unsigned stride0)
{
    unsigned long long ga = (unsigned long long)(uintptr_t)gaddr;
    u32x4 g0;
    g0.x = 1u;                                           // count=1, user mode
    g0.y = lds_addr;                                     // bits 63:32 lds_addr
    g0.z = (unsigned)(ga & 0xFFFFFFFFull);               // global_addr[31:0]
    g0.w = (unsigned)((ga >> 32) & 0x1FFFFFFull)         // global_addr[56:32]
         | (2u << 30);                                   // type = 2 ("image")
    i32x8 g1;
    g1[0] = (int)((1u << 16)                             // data_size = 1 (2 bytes)
                | (1u << 20)                             // pad_enable
                | (3u << 22)                             // pad_interval: 16 DWORDs
                | (3u << 25));                           // pad_amount: 4 DWORDs
    g1[1] = (int)((tensor_d0 & 0xFFFFu) << 16);          // tensor_dim0[15:0] @ 63:48
    g1[2] = (int)(((tensor_d0 >> 16) & 0xFFFFu)          // tensor_dim0[31:16]
                | ((tensor_d1 & 0xFFFFu) << 16));        // tensor_dim1[15:0]
    g1[3] = (int)(((tensor_d1 >> 16) & 0xFFFFu)          // tensor_dim1[31:16]
                | ((tile_d0 & 0xFFFFu) << 16));          // tile_dim0
    g1[4] = (int)(tile_d1 & 0xFFFFu);                    // tile_dim1 (tile_dim2 = 0)
    g1[5] = (int)stride0;                                // tensor_dim0_stride[31:0]
    g1[6] = 0;                                           // stride hi / dim1_stride lo
    g1[7] = 0;
    i32x4 gz = {0, 0, 0, 0};                             // 2D: groups 2/3 unused
    i32x8 gz8 = {0, 0, 0, 0, 0, 0, 0, 0};
    __builtin_amdgcn_tensor_load_to_lds(g0, g1, gz, gz, gz8, 0);
}
#endif

// ---------------------------------------------------------------------------
// Elementwise helpers
// ---------------------------------------------------------------------------
__global__ void k_f2bf(const float* __restrict__ src, bf16_t* __restrict__ dst, int n) {
    int i = blockIdx.x * blockDim.x + threadIdx.x;
    if (i < n) dst[i] = f2bf(src[i]);
}

__global__ void k_zero(float* __restrict__ p, int n) {
    int i = blockIdx.x * blockDim.x + threadIdx.x;
    if (i < n) p[i] = 0.f;
}

// ---------------------------------------------------------------------------
// WMMA bf16 GEMM:  C(M x N) = A(M x K) * W(N x K)^T,  K == 256 fixed.
// Block tile 128(M) x 64(N), 256 threads = 8 wave32; per-wave 16x64 strip =
// 4 accumulators. K in 32-wide steps, LDS double-buffered; tiles staged by
// the Tensor Data Mover when available.
// ---------------------------------------------------------------------------
enum { EP_BF16 = 0, EP_F32 = 1, EP_MEAN = 2, EP_HEADX = 3, EP_MU = 4, EP_SIGMA = 5 };

template <int EP>
__global__ __launch_bounds__(256)
void k_gemm(const bf16_t* __restrict__ A, const bf16_t* __restrict__ W,
            int N, int ldc,
            bf16_t* __restrict__ outBf, float* __restrict__ outF,
            const float* __restrict__ bias, bf16_t* __restrict__ out2Bf)
{
    constexpr int K   = 256;
    constexpr int NK  = 8;    // K / 32
    constexpr int LDA = 40;   // LDS row stride (elems): 80B rows, 16B aligned chunks

    __shared__ __attribute__((aligned(16))) bf16_t As[2][128 * LDA];
    __shared__ __attribute__((aligned(16))) bf16_t Bs[2][64 * LDA];

    const int tid  = threadIdx.x;
    const int lane = tid & 31;
    const int wv   = tid >> 5;
    const int m0   = blockIdx.y * 128;
    const int n0   = blockIdx.x * 64;

    v8f acc[4];
#pragma unroll
    for (int i = 0; i < 4; ++i)
#pragma unroll
        for (int j = 0; j < 8; ++j) acc[i][j] = 0.f;

    const int arow = wv * 16 + (lane & 15);
    const int koff = (lane >> 4) * 8;   // ISA 7.12.2: lanes 0-15 K{0..7,16..23}; 16-31 K{8..15,24..31}

    auto compute = [&](int buf) {
        v16bf af;
        v8bf alo = *(const v8bf*)&As[buf][arow * LDA + koff];
        v8bf ahi = *(const v8bf*)&As[buf][arow * LDA + 16 + koff];
#pragma unroll
        for (int i = 0; i < 8; ++i) { af[i] = alo[i]; af[8 + i] = ahi[i]; }
#pragma unroll
        for (int nt = 0; nt < 4; ++nt) {
            int bcol = nt * 16 + (lane & 15);
            v8bf blo = *(const v8bf*)&Bs[buf][bcol * LDA + koff];
            v8bf bhi = *(const v8bf*)&Bs[buf][bcol * LDA + 16 + koff];
            v16bf bfr;
#pragma unroll
            for (int i = 0; i < 8; ++i) { bfr[i] = blo[i]; bfr[8 + i] = bhi[i]; }
            acc[nt] = __builtin_amdgcn_wmma_f32_16x16x32_bf16(
                false, af, false, bfr, (short)0, acc[nt], false, false);
        }
    };

#if USE_TDM
    // ---- Tensor Data Mover staging: one wave drives both tile DMAs ----
    const unsigned Mtot = (unsigned)gridDim.y * 128u;
    auto issue = [&](int buf, int k0) {
        if (wv == 0) {
            tdm_load_2d_bf16((unsigned)(uintptr_t)(void*)&As[buf][0],
                             (const void*)(A + (size_t)m0 * K + k0),
                             (unsigned)K, Mtot, 32u, 128u, (unsigned)K);
            tdm_load_2d_bf16((unsigned)(uintptr_t)(void*)&Bs[buf][0],
                             (const void*)(W + (size_t)n0 * K + k0),
                             (unsigned)K, (unsigned)N, 32u, 64u, (unsigned)K);
        }
    };
    issue(0, 0);
    if (wv == 0) __builtin_amdgcn_s_wait_tensorcnt(0);
    __syncthreads();
    int cur = 0;
    for (int ks = 0; ks < NK; ++ks) {
        if (ks + 2 < NK)
            __builtin_prefetch((const void*)(A + (size_t)(m0 + (tid >> 2)) * K + (ks + 2) * 32), 0, 1);
        if (ks + 1 < NK) issue(cur ^ 1, (ks + 1) * 32);
        compute(cur);
        if (wv == 0) __builtin_amdgcn_s_wait_tensorcnt(0);
        __syncthreads();
        cur ^= 1;
    }
#elif USE_ASYNC
    // ---- per-lane async global->LDS staging (ASYNCcnt) ----
    if (n0 + 64 > N) {  // zero-fill B tail once (async path can't pad)
        for (int i = tid; i < 64 * LDA; i += 256) { Bs[0][i] = (bf16_t)0.f; Bs[1][i] = (bf16_t)0.f; }
    }
    __syncthreads();
    auto issue = [&](int buf, int k0) {
        { int row = tid >> 2, seg = tid & 3;
          GASYNC_B128(A + (size_t)(m0 + row) * K + k0 + seg * 8,
                      &As[buf][row * LDA + seg * 8]); }
        { int c = tid + 256; int row = c >> 2, seg = c & 3;
          GASYNC_B128(A + (size_t)(m0 + row) * K + k0 + seg * 8,
                      &As[buf][row * LDA + seg * 8]); }
        { int row = tid >> 2, seg = tid & 3; int col = n0 + row;
          if (col < N)
              GASYNC_B128(W + (size_t)col * K + k0 + seg * 8,
                          &Bs[buf][row * LDA + seg * 8]); }
    };
    issue(0, 0);
    __builtin_amdgcn_s_wait_asynccnt(0);
    __syncthreads();
    int cur = 0;
    for (int ks = 0; ks < NK; ++ks) {
        if (ks + 1 < NK) issue(cur ^ 1, (ks + 1) * 32);
        compute(cur);
        __builtin_amdgcn_s_wait_asynccnt(0);
        __syncthreads();
        cur ^= 1;
    }
#else
    // ---- fallback: global->VGPR->ds_store double buffering ----
    auto ldA = [&](int c, int k0) -> u32x4 {
        int row = c >> 2, seg = c & 3;
        return *(const u32x4*)(A + (size_t)(m0 + row) * K + k0 + seg * 8);
    };
    auto ldB = [&](int c, int k0) -> u32x4 {
        int row = c >> 2, seg = c & 3;
        int col = n0 + row;
        if (col < N) return *(const u32x4*)(W + (size_t)col * K + k0 + seg * 8);
        u32x4 z = {0u, 0u, 0u, 0u};
        return z;
    };
    auto stAB = [&](int buf, u32x4 a0, u32x4 a1, u32x4 b0) {
        { int row = tid >> 2, seg = tid & 3;
          *(u32x4*)&As[buf][row * LDA + seg * 8] = a0; }
        { int c = tid + 256; int row = c >> 2, seg = c & 3;
          *(u32x4*)&As[buf][row * LDA + seg * 8] = a1; }
        { int row = tid >> 2, seg = tid & 3;
          *(u32x4*)&Bs[buf][row * LDA + seg * 8] = b0; }
    };
    u32x4 a0 = ldA(tid, 0), a1 = ldA(tid + 256, 0), b0 = ldB(tid, 0);
    stAB(0, a0, a1, b0);
    __syncthreads();
    int cur = 0;
    for (int ks = 0; ks < NK; ++ks) {
        if (ks + 2 < NK)
            __builtin_prefetch((const void*)(A + (size_t)(m0 + (tid >> 2)) * K + (ks + 2) * 32), 0, 1);
        if (ks + 1 < NK) {
            int k0 = (ks + 1) * 32;
            a0 = ldA(tid, k0); a1 = ldA(tid + 256, k0); b0 = ldB(tid, k0);
            stAB(cur ^ 1, a0, a1, b0);
        }
        compute(cur);
        __syncthreads();
        cur ^= 1;
    }
#endif

    // --- epilogue: C/D layout: VGPR r, lanes 0-15 -> M=r, lanes 16-31 -> M=8+r
    const int rowBase = m0 + wv * 16 + ((lane >> 4) << 3);
    const int colLane = lane & 15;
#pragma unroll
    for (int nt = 0; nt < 4; ++nt) {
        int col = n0 + nt * 16 + colLane;
        if (col >= N) continue;
#pragma unroll
        for (int r = 0; r < 8; ++r) {
            int row = rowBase + r;
            float v = acc[nt][r];
            if constexpr (EP == EP_BF16) {
                outBf[(size_t)row * ldc + col] = f2bf(v);
            } else if constexpr (EP == EP_F32) {
                outF[(size_t)row * ldc + col] = v;
            } else if constexpr (EP == EP_MEAN) {
                atomicAdd(&outF[(size_t)(row >> 7) * ldc + col], v * 0.0078125f);
            } else if constexpr (EP == EP_HEADX) {
                float t = tanhf(v + bias[col]);
                float x = t > 0.f ? t : (__expf(t) - 1.f);   // elu(tanh(.))
                outF[(size_t)row * ldc + col]   = x;
                out2Bf[(size_t)row * ldc + col] = f2bf(x);
            } else if constexpr (EP == EP_MU) {
                outF[(size_t)row * ldc + col] = v + bias[col];
            } else {  // EP_SIGMA
                float s = v + bias[col];
                float e = s > 0.f ? s : (__expf(s) - 1.f);
                outF[(size_t)row * ldc + col] = e + 1.0f + 1e-14f;
            }
        }
    }
}

// ---------------------------------------------------------------------------
// Depthwise causal conv (k=4) over time + SiLU.
// ---------------------------------------------------------------------------
__global__ __launch_bounds__(256)
void k_conv_silu(const bf16_t* __restrict__ xz, const float* __restrict__ cw,
                 const float* __restrict__ cb, bf16_t* __restrict__ xc)
{
    int idx = blockIdx.x * 256 + threadIdx.x;   // m * 256 + d
    int d = idx & 255;
    int m = idx >> 8;
    int t = m & 127;                            // L = 128
    float a = cb[d];
#pragma unroll
    for (int k = 0; k < 4; ++k) {
        int tt = t - 3 + k;
        if (tt >= 0) a += cw[d * 4 + k] * bf2f(xz[(size_t)(m - 3 + k) * 512 + d]);
    }
    float s = a * (1.f / (1.f + __expf(-a)));   // silu
    xc[idx] = f2bf(s);
}

// ---------------------------------------------------------------------------
// Selective scan, fused dt_proj + softplus + D-skip + SiLU(z) gating.
// x_dbl row staged to LDS via async global->LDS loads when available.
// ---------------------------------------------------------------------------
__global__ __launch_bounds__(256)
void k_scan(const bf16_t* __restrict__ xc, const bf16_t* __restrict__ xz,
            const float* __restrict__ xdbl, const float* __restrict__ wdtp,
            const float* __restrict__ dtb, const float* __restrict__ A_log,
            const float* __restrict__ Dskip, bf16_t* __restrict__ yf)
{
    const int b = blockIdx.x;
    const int d = threadIdx.x;
    __shared__ __attribute__((aligned(16))) float s[48];

    float wdt[16], nA[16], h[16];
#pragma unroll
    for (int r = 0; r < 16; ++r) wdt[r] = wdtp[d * 16 + r];
#pragma unroll
    for (int n = 0; n < 16; ++n) { nA[n] = -__expf(A_log[d * 16 + n]); h[n] = 0.f; }
    const float bias = dtb[d];
    const float dsk  = Dskip[d];

    for (int t = 0; t < 128; ++t) {
        const size_t m = (size_t)b * 128 + t;
#if USE_ASYNC
        if (threadIdx.x < 12)
            GASYNC_B128(xdbl + m * 48 + threadIdx.x * 4, &s[threadIdx.x * 4]);
        __builtin_amdgcn_s_wait_asynccnt(0);
#else
        if (threadIdx.x < 48) s[threadIdx.x] = xdbl[m * 48 + threadIdx.x];
#endif
        __syncthreads();

        float dtr = bias;
#pragma unroll
        for (int r = 0; r < 16; ++r) dtr += wdt[r] * s[r];
        float dt = dtr > 20.f ? dtr : log1pf(__expf(dtr));   // softplus
        float xt = bf2f(xc[m * 256 + d]);
        float dx = dt * xt;
        float y = 0.f;
#pragma unroll
        for (int n = 0; n < 16; ++n) {
            float dA = __expf(dt * nA[n]);
            h[n] = dA * h[n] + dx * s[16 + n];
            y += h[n] * s[32 + n];
        }
        float z = bf2f(xz[m * 512 + 256 + d]);
        float g = z * (1.f / (1.f + __expf(-z)));            // silu(z)
        yf[m * 256 + d] = f2bf((y + xt * dsk) * g);
        __syncthreads();
    }
}

// ---------------------------------------------------------------------------
// Host launch
// ---------------------------------------------------------------------------
extern "C" void kernel_launch(void* const* d_in, const int* in_sizes, int n_in,
                              void* d_out, int out_size, void* d_ws, size_t ws_size,
                              hipStream_t stream)
{
    (void)in_sizes; (void)n_in; (void)out_size; (void)ws_size;
    const int    Bn = 2048, L = 128;
    const size_t M  = (size_t)Bn * L;   // 262144

    const float* input     = (const float*)d_in[0];
    // d_in[1] = edge_index (unused by reference)
    const float* in_proj_w = (const float*)d_in[2];
    const float* conv_w    = (const float*)d_in[3];
    const float* conv_b    = (const float*)d_in[4];
    const float* x_proj_w  = (const float*)d_in[5];
    const float* dt_proj_w = (const float*)d_in[6];
    const float* dt_proj_b = (const float*)d_in[7];
    const float* A_log     = (const float*)d_in[8];
    const float* D_skip    = (const float*)d_in[9];
    const float* out_proj_w= (const float*)d_in[10];
    const float* out_fc_w  = (const float*)d_in[11];
    const float* out_fc_b  = (const float*)d_in[12];
    const float* mu_fc_w   = (const float*)d_in[13];
    const float* mu_fc_b   = (const float*)d_in[14];
    const float* sigma_fc_w= (const float*)d_in[15];
    const float* sigma_fc_b= (const float*)d_in[16];

    char* ws = (char*)d_ws;
    size_t off = 0;
    auto alloc = [&](size_t bytes) -> char* {
        char* p = ws + off;
        off += (bytes + 255) & ~(size_t)255;
        return p;
    };
    bf16_t* xbf  = (bf16_t*)alloc(M * 256 * 2);
    bf16_t* xzbf = (bf16_t*)alloc(M * 512 * 2);
    bf16_t* xcbf = (bf16_t*)alloc(M * 256 * 2);
    float*  xdbl = (float* )alloc(M * 48 * 4);
    bf16_t* yfbf = (bf16_t*)alloc(M * 256 * 2);
    float*  ef   = (float* )alloc((size_t)Bn * 256 * 4);
    bf16_t* ebf  = (bf16_t*)alloc((size_t)Bn * 256 * 2);
    bf16_t* xhbf = (bf16_t*)alloc((size_t)Bn * 256 * 2);
    bf16_t* wIn  = (bf16_t*)alloc(512 * 256 * 2);
    bf16_t* wXp  = (bf16_t*)alloc(48 * 256 * 2);
    bf16_t* wOp  = (bf16_t*)alloc(256 * 256 * 2);
    bf16_t* wFc  = (bf16_t*)alloc(256 * 256 * 2);
    bf16_t* wMu  = (bf16_t*)alloc(64 * 256 * 2);
    bf16_t* wSg  = (bf16_t*)alloc(64 * 256 * 2);

    auto cvt = [&](const float* s, bf16_t* dst, size_t n) {
        k_f2bf<<<dim3((unsigned)((n + 255) / 256)), dim3(256), 0, stream>>>(s, dst, (int)n);
    };
    cvt(input,      xbf, M * 256);
    cvt(in_proj_w,  wIn, 512 * 256);
    cvt(x_proj_w,   wXp, 48 * 256);
    cvt(out_proj_w, wOp, 256 * 256);
    cvt(out_fc_w,   wFc, 256 * 256);
    cvt(mu_fc_w,    wMu, 64 * 256);
    cvt(sigma_fc_w, wSg, 64 * 256);
    k_zero<<<dim3((unsigned)(Bn * 256 / 256)), dim3(256), 0, stream>>>(ef, Bn * 256);

    // xz = x @ in_proj_w^T : (M, 512)
    k_gemm<EP_BF16><<<dim3(8, (unsigned)(M / 128)), dim3(256), 0, stream>>>(
        xbf, wIn, 512, 512, xzbf, nullptr, nullptr, nullptr);

    // depthwise causal conv + silu -> xc : (M, 256)
    k_conv_silu<<<dim3((unsigned)M), dim3(256), 0, stream>>>(xzbf, conv_w, conv_b, xcbf);

    // x_dbl = xc @ x_proj_w^T : (M, 48) f32
    k_gemm<EP_F32><<<dim3(1, (unsigned)(M / 128)), dim3(256), 0, stream>>>(
        xcbf, wXp, 48, 48, nullptr, xdbl, nullptr, nullptr);

    // selective scan + gating -> yf : (M, 256) bf16
    k_scan<<<dim3(Bn), dim3(256), 0, stream>>>(
        xcbf, xzbf, xdbl, dt_proj_w, dt_proj_b, A_log, D_skip, yfbf);

    // e = mean_t(yf @ out_proj_w^T) : (Bn, 256) via atomic accumulation
    k_gemm<EP_MEAN><<<dim3(4, (unsigned)(M / 128)), dim3(256), 0, stream>>>(
        yfbf, wOp, 256, 256, nullptr, ef, nullptr, nullptr);

    cvt(ef, ebf, (size_t)Bn * 256);

    float* out_x  = (float*)d_out;
    float* out_mu = out_x + (size_t)Bn * 256;
    float* out_sg = out_mu + (size_t)Bn * 64;

    // x = elu(tanh(e @ out_fc_w^T + b)) -> out_x (f32) and xhbf (bf16)
    k_gemm<EP_HEADX><<<dim3(4, Bn / 128), dim3(256), 0, stream>>>(
        ebf, wFc, 256, 256, nullptr, out_x, out_fc_b, xhbf);

    // mu = x @ mu_fc_w^T + b
    k_gemm<EP_MU><<<dim3(1, Bn / 128), dim3(256), 0, stream>>>(
        xhbf, wMu, 64, 64, nullptr, out_mu, mu_fc_b, nullptr);

    // sigma = elu(x @ sigma_fc_w^T + b) + 1 + 1e-14
    k_gemm<EP_SIGMA><<<dim3(1, Bn / 128), dim3(256), 0, stream>>>(
        xhbf, wSg, 64, 64, nullptr, out_sg, sigma_fc_b, nullptr);
}
